// b_RGLSTM_15272903704605
// MI455X (gfx1250) — matrix operand order repaired and checked by reading, not compiled, not run on hardware
//
#include <hip/hip_runtime.h>
#include <hip/hip_bf16.h>

// ---------------- problem dims ----------------
#define B_   64
#define S_   512
#define D_   512
#define H_   512
#define G4   2048   // 4*H
#define NH2  1024   // 2*H
#define NWG  32     // persistent recurrence workgroups (H/16)

typedef __bf16 bf16_t;
typedef __attribute__((ext_vector_type(16))) __bf16 v16bf;
typedef __attribute__((ext_vector_type(8)))  float  v8f;
typedef __attribute__((ext_vector_type(4)))  unsigned int v4u;
typedef __attribute__((ext_vector_type(8)))  int  v8i;
typedef __attribute__((ext_vector_type(4)))  int  v4i;

// output-buffer layout (flat concat, element offsets, fp32)
#define OFF_HID  ((size_t)0)                               // hidden_seq (B,S,H)
#define N_HID    ((size_t)B_ * S_ * H_)
#define N_MU     ((size_t)B_ * (S_ + 1) * H_)
#define OFF_HMU  (OFF_HID + N_HID)
#define OFF_HSD  (OFF_HMU + N_MU)
#define OFF_CMU  (OFF_HSD + N_MU)
#define OFF_CSD  (OFF_CMU + N_MU)
#define OFF_HT   (OFF_CSD + N_MU)
#define OFF_CT   (OFF_HT + (size_t)B_ * H_)

// dynamic LDS layout (bytes); dynamic LDS base offset is 0 (no static LDS here)
#define LDS_WHH   0         // bf16[4][16][512] : W_hh gate-row slices      (64 KB)
#define LDS_WG    65536     // bf16[4][16][512] : Whg/Wcg mu/std row slices (64 KB)
#define LDS_ST0   131072    // bf16[64][512]    : staged h / h_new          (64 KB)
#define LDS_ST1   196608    // bf16[64][512]    : staged c_new              (64 KB)
#define LDS_GSH   262144    // float[4][64][16] : gate/p tile exchange      (16 KB)
#define LDS_TOTAL 278528

#if __has_builtin(__builtin_amdgcn_tensor_load_to_lds) && \
    __has_builtin(__builtin_amdgcn_s_wait_tensorcnt)
#define USE_TDM 1
#else
#define USE_TDM 0
#endif

// ---------------- WMMA helpers ----------------
__device__ __forceinline__ v8f wmma_bf16(v16bf a, v16bf b, v8f c) {
  return __builtin_amdgcn_wmma_f32_16x16x32_bf16(false, a, false, b, (short)0, c, false, false);
}

// A fragment: 16x32 tile, row-major A[m][k], ld = row stride (elems).
__device__ __forceinline__ v16bf load_frag_a(const bf16_t* tile, int ld) {
  int lane = threadIdx.x & 31;
  int m  = lane & 15;
  int kb = (lane >> 4) << 3;
  const bf16_t* p = tile + (size_t)m * ld + kb;
  union { uint4 u[2]; v16bf v; } r;
  r.u[0] = *(const uint4*)(p);
  r.u[1] = *(const uint4*)(p + 16);
  return r.v;
}

// B fragment: 32x16 (KxN), B[k][n] = W[n][k] with W row-major [n][k], ld = K stride.
__device__ __forceinline__ v16bf load_frag_b(const bf16_t* wtile, int ld) {
  int lane = threadIdx.x & 31;
  int n  = lane & 15;
  int kb = (lane >> 4) << 3;
  const bf16_t* p = wtile + (size_t)n * ld + kb;
  union { uint4 u[2]; v16bf v; } r;
  r.u[0] = *(const uint4*)(p);
  r.u[1] = *(const uint4*)(p + 16);
  return r.v;
}

__device__ __forceinline__ float sigmoidf_(float x) { return 1.0f / (1.0f + __expf(-x)); }
__device__ __forceinline__ float softplusf_(float x) {
  return fmaxf(x, 0.0f) + log1pf(__expf(-fabsf(x)));
}
__device__ __forceinline__ float clampf_(float x, float lo, float hi) {
  return fminf(fmaxf(x, lo), hi);
}

__device__ __forceinline__ void copy_lds_u4(bf16_t* dst, const bf16_t* src, int n_u4) {
  uint4* d = (uint4*)dst;
  const uint4* s = (const uint4*)src;
  for (int i = threadIdx.x; i < n_u4; i += 256) d[i] = s[i];
}

#if USE_TDM
// Tensor Data Mover: one linear 1-D tile (bytes/8 elements of 8 B), global -> LDS.
// D# per CDNA5 ISA. Issued by one wave (TDM ignores EXEC).
__device__ __forceinline__ void tdm_load_linear(unsigned lds_off, const void* gptr,
                                                unsigned bytes) {
  unsigned long long ga = (unsigned long long)gptr;
  unsigned n8 = bytes >> 3;                       // elements of data_size = 8 B
  v4u g0;
  g0[0] = 1u;                                     // count=1, user descriptor
  g0[1] = lds_off;                                // lds_addr (bytes)
  g0[2] = (unsigned)ga;                           // global_addr[31:0]
  g0[3] = (unsigned)(ga >> 32) | (2u << 30);      // global_addr[56:32] | type=2
  v8i g1;
  g1[0] = (int)(3u << 16);                        // workgroup_mask=0, data_size=3 (8 B)
  g1[1] = (int)((n8 & 0xFFFFu) << 16);            // tensor_dim0[15:0]
  g1[2] = (int)((n8 >> 16) | (1u << 16));         // tensor_dim0[31:16] | tensor_dim1=1
  g1[3] = (int)(n8 << 16);                        // tile_dim0 (16-bit)
  g1[4] = 1;                                      // tile_dim1=1, tile_dim2=0
  g1[5] = (int)n8;                                // tensor_dim0_stride[31:0]
  g1[6] = (int)((n8 & 0xFFFFu) << 16);            // stride0[47:32]=0 | stride1[15:0]
  g1[7] = 0;
  v4i z4 = {0, 0, 0, 0};
#if defined(__clang_major__) && (__clang_major__ >= 23)
  v8i z8 = {0, 0, 0, 0, 0, 0, 0, 0};
  __builtin_amdgcn_tensor_load_to_lds(g0, g1, z4, z4, z8, 0);
#else
  __builtin_amdgcn_tensor_load_to_lds(g0, g1, z4, z4, 0);
#endif
}
#endif

// Stage a contiguous global block into LDS; TDM path when available.
__device__ __forceinline__ void stage_to_lds(unsigned lds_off, bf16_t* lds_dst,
                                             const bf16_t* gsrc, unsigned bytes) {
#if USE_TDM
  if (threadIdx.x < 32) tdm_load_linear(lds_off, gsrc, bytes);
  (void)lds_dst;
#else
  (void)lds_off;
  copy_lds_u4(lds_dst, gsrc, bytes >> 4);
#endif
}
__device__ __forceinline__ void stage_wait() {
#if USE_TDM
  if (threadIdx.x < 32) __builtin_amdgcn_s_wait_tensorcnt((short)0);
#endif
  __syncthreads();
}

// device-wide sense-reversal barrier for the persistent kernel (NWG blocks).
// Also issues the CDNA5 cluster barrier (S_NOP when not launched as a cluster).
__device__ __forceinline__ void grid_barrier(unsigned* bar) {
  __threadfence();
  __syncthreads();
  __builtin_amdgcn_s_cluster_barrier();
  if (threadIdx.x == 0) {
    unsigned g = __hip_atomic_load(&bar[1], __ATOMIC_ACQUIRE, __HIP_MEMORY_SCOPE_AGENT);
    unsigned old = __hip_atomic_fetch_add(&bar[0], 1u, __ATOMIC_ACQ_REL, __HIP_MEMORY_SCOPE_AGENT);
    if (old == NWG - 1) {
      __hip_atomic_store(&bar[0], 0u, __ATOMIC_RELAXED, __HIP_MEMORY_SCOPE_AGENT);
      __hip_atomic_fetch_add(&bar[1], 1u, __ATOMIC_RELEASE, __HIP_MEMORY_SCOPE_AGENT);
    } else {
      while (__hip_atomic_load(&bar[1], __ATOMIC_ACQUIRE, __HIP_MEMORY_SCOPE_AGENT) == g)
        __builtin_amdgcn_s_sleep(2);
    }
  }
  __syncthreads();
}

// ---------------- conversion kernels ----------------
__global__ void cvt_bf16_kernel(const float* __restrict__ src, bf16_t* __restrict__ dst, int n) {
  int i = blockIdx.x * blockDim.x + threadIdx.x;
  if (i < n) dst[i] = (bf16_t)src[i];
}

// W_hg/W_cg are (H, 2H); store Wᵀ[n][k] = W[k][n], (2H, H) row-major bf16.
__global__ void cvt_T_kernel(const float* __restrict__ src, bf16_t* __restrict__ dstT) {
  int i = blockIdx.x * blockDim.x + threadIdx.x;
  if (i < NH2 * H_) {
    int n = i / H_, k = i % H_;
    dstT[i] = (bf16_t)src[(size_t)k * NH2 + n];
  }
}

__global__ void bias_kernel(const float* __restrict__ bih, const float* __restrict__ bhh,
                            float* __restrict__ bias) {
  int i = blockIdx.x * blockDim.x + threadIdx.x;
  if (i < G4) bias[i] = bih[i] + bhh[i];
}

// h0/c0 from _sample(zeros): mu = 1e-6, std = softplus(0) = ln2. Also resets barrier state.
__global__ void init_kernel(const float* __restrict__ eps_h0, const float* __restrict__ eps_c0,
                            bf16_t* __restrict__ h_bf, float* __restrict__ c_buf,
                            float* __restrict__ out, unsigned* __restrict__ bar) {
  int e = blockIdx.x * blockDim.x + threadIdx.x;
  if (e == 0) { bar[0] = 0u; bar[1] = 0u; }
  if (e >= B_ * H_) return;
  int b = e >> 9, j = e & (H_ - 1);
  const float mu0 = 1e-6f;
  const float sd0 = 0.6931471805599453f;
  h_bf[e] = (bf16_t)(mu0 + eps_h0[e] * sd0);
  c_buf[e] = mu0 + eps_c0[e] * sd0;
  size_t o = (size_t)b * (S_ + 1) * H_ + j;   // t = 0 slot
  out[OFF_HMU + o] = mu0;
  out[OFF_HSD + o] = sd0;
  out[OFF_CMU + o] = mu0;
  out[OFF_CSD + o] = sd0;
}

// ---------------- big parallel GEMM: Gx = X @ W_ihᵀ + (b_ih + b_hh) ----------------
// M = B*S = 32768, N = 2048, K = 512. Block tile 64x128, 8 waves (2x4), wave tile 32x32.
__global__ __launch_bounds__(256) void gemm_gx_kernel(const bf16_t* __restrict__ X,
                                                      const bf16_t* __restrict__ Wih,
                                                      const float* __restrict__ bias,
                                                      float* __restrict__ Gx) {
  int wave = threadIdx.x >> 5;
  int wm = wave & 1, wn = wave >> 1;
  int m0 = blockIdx.y * 64 + wm * 32;
  int n0 = blockIdx.x * 128 + wn * 32;
  v8f acc[2][2] = {};
  for (int kb = 0; kb < D_; kb += 32) {
    __builtin_prefetch(Wih + (size_t)n0 * D_ + kb + 64, 0, 1);  // global_prefetch_b8
    v16bf a0 = load_frag_a(X + (size_t)m0 * D_ + kb, D_);
    v16bf a1 = load_frag_a(X + (size_t)(m0 + 16) * D_ + kb, D_);
    v16bf b0 = load_frag_b(Wih + (size_t)n0 * D_ + kb, D_);
    v16bf b1 = load_frag_b(Wih + (size_t)(n0 + 16) * D_ + kb, D_);
    acc[0][0] = wmma_bf16(a0, b0, acc[0][0]);
    acc[0][1] = wmma_bf16(a0, b1, acc[0][1]);
    acc[1][0] = wmma_bf16(a1, b0, acc[1][0]);
    acc[1][1] = wmma_bf16(a1, b1, acc[1][1]);
  }
  int lane = threadIdx.x & 31;
  int nlo = lane & 15, mh = (lane >> 4) * 8;
#pragma unroll
  for (int mi = 0; mi < 2; ++mi)
#pragma unroll
    for (int ni = 0; ni < 2; ++ni)
#pragma unroll
      for (int r = 0; r < 8; ++r) {
        int row = m0 + mi * 16 + mh + r;
        int col = n0 + ni * 16 + nlo;
        Gx[(size_t)row * G4 + col] = acc[mi][ni][r] + bias[col];
      }
}

// ---------------- persistent recurrence kernel ----------------
// NWG = 32 workgroups, each owning H-columns [j0, j0+16). One launch covers all 512 steps.
// Weight slices live in LDS for the whole kernel; activations staged per phase via TDM.
__global__ __launch_bounds__(256) void recurrence_kernel(
    const bf16_t* __restrict__ Whh, const bf16_t* __restrict__ WhgT,
    const bf16_t* __restrict__ WcgT, const float* __restrict__ Gx,
    const float* __restrict__ eps_h, const float* __restrict__ eps_c,
    bf16_t* __restrict__ h_bf, bf16_t* __restrict__ hn_bf, bf16_t* __restrict__ cn_bf,
    float* __restrict__ c_buf, float* __restrict__ out, unsigned* __restrict__ bar) {
  extern __shared__ char smem[];
  bf16_t* whh_sl = (bf16_t*)(smem + LDS_WHH);
  bf16_t* wg_sl  = (bf16_t*)(smem + LDS_WG);
  bf16_t* st0    = (bf16_t*)(smem + LDS_ST0);
  bf16_t* st1    = (bf16_t*)(smem + LDS_ST1);
  float*  gsh    = (float*)(smem + LDS_GSH);

  int j0 = blockIdx.x * 16;
  int wave = threadIdx.x >> 5;
  int q = wave >> 1;                 // phase A: gate 0..3 ; phase B: {h-mu,h-std,c-mu,c-std}
  int mh2 = (wave & 1) * 32;         // batch-row half
  int lane = threadIdx.x & 31;
  int nlo = lane & 15, mh = (lane >> 4) * 8;

  // ---- stage weight slices into LDS once: 8 contiguous 16 KB blocks ----
  for (int g = 0; g < 4; ++g) {
    stage_to_lds(LDS_WHH + g * 16384, whh_sl + g * 16 * H_,
                 Whh + (size_t)(g * H_ + j0) * H_, 16384);
    const bf16_t* WT = (g < 2) ? WhgT : WcgT;
    stage_to_lds(LDS_WG + g * 16384, wg_sl + g * 16 * H_,
                 WT + (size_t)((g & 1) * H_ + j0) * H_, 16384);
  }
  stage_wait();

  for (int t = 0; t < S_; ++t) {
    // ================= phase A: gates = h @ W_hhᵀ + Gx[t] ; LSTM cell =================
    stage_to_lds(LDS_ST0, st0, h_bf, 65536);     // stage h (64x512 bf16)
    stage_wait();
    {
      v8f acc[2] = {};
      const bf16_t* wrow = whh_sl + (size_t)q * 16 * H_;
#pragma unroll 4
      for (int kb = 0; kb < H_; kb += 32) {
        v16bf a0 = load_frag_a(st0 + (size_t)mh2 * H_ + kb, H_);
        v16bf a1 = load_frag_a(st0 + (size_t)(mh2 + 16) * H_ + kb, H_);
        v16bf b = load_frag_b(wrow + kb, H_);
        acc[0] = wmma_bf16(a0, b, acc[0]);
        acc[1] = wmma_bf16(a1, b, acc[1]);
      }
#pragma unroll
      for (int mi = 0; mi < 2; ++mi)
#pragma unroll
        for (int r = 0; r < 8; ++r)
          gsh[((q * 64) + mh2 + mi * 16 + mh + r) * 16 + nlo] = acc[mi][r];
    }
    __syncthreads();
    for (int e = threadIdx.x; e < 64 * 16; e += 256) {
      int b = e >> 4, n = e & 15, j = j0 + n;
      const float* gx = Gx + ((size_t)b * S_ + t) * G4 + j;
      float iv = sigmoidf_(gsh[(0 * 64 + b) * 16 + n] + gx[0]);
      float fv = sigmoidf_(gsh[(1 * 64 + b) * 16 + n] + gx[512]);
      float gv = tanhf(gsh[(2 * 64 + b) * 16 + n] + gx[1024]);
      float ov = sigmoidf_(gsh[(3 * 64 + b) * 16 + n] + gx[1536]);
      float cp = c_buf[b * H_ + j];
      float cn = fv * cp + iv * gv;
      float hn = ov * tanhf(cn);
      hn_bf[b * H_ + j] = (bf16_t)hn;
      cn_bf[b * H_ + j] = (bf16_t)cn;
    }
    grid_barrier(bar);

    // ================= phase B: sampling GEMMs + reparameterize =================
    stage_to_lds(LDS_ST0, st0, hn_bf, 65536);    // stage h_new
    stage_to_lds(LDS_ST1, st1, cn_bf, 65536);    // stage c_new
    stage_wait();
    {
      v8f acc[2] = {};
      const bf16_t* A = (q < 2) ? st0 : st1;
      const bf16_t* wrow = wg_sl + (size_t)q * 16 * H_;
#pragma unroll 4
      for (int kb = 0; kb < H_; kb += 32) {
        v16bf a0 = load_frag_a(A + (size_t)mh2 * H_ + kb, H_);
        v16bf a1 = load_frag_a(A + (size_t)(mh2 + 16) * H_ + kb, H_);
        v16bf b = load_frag_b(wrow + kb, H_);
        acc[0] = wmma_bf16(a0, b, acc[0]);
        acc[1] = wmma_bf16(a1, b, acc[1]);
      }
#pragma unroll
      for (int mi = 0; mi < 2; ++mi)
#pragma unroll
        for (int r = 0; r < 8; ++r)
          gsh[((q * 64) + mh2 + mi * 16 + mh + r) * 16 + nlo] = acc[mi][r];
    }
    __syncthreads();
    for (int e = threadIdx.x; e < 64 * 16; e += 256) {
      int b = e >> 4, n = e & 15, j = j0 + n;
      float mu_h = clampf_(gsh[(0 * 64 + b) * 16 + n], 1e-6f, 1e6f);
      float sd_h = fmaxf(softplusf_(gsh[(1 * 64 + b) * 16 + n]), 1e-6f);
      float mu_c = clampf_(gsh[(2 * 64 + b) * 16 + n], 1e-6f, 1e6f);
      float sd_c = fmaxf(softplusf_(gsh[(3 * 64 + b) * 16 + n]), 1e-6f);
      size_t eo = ((size_t)(t + 1) * B_ + b) * H_ + j;
      float hs = mu_h + eps_h[eo] * sd_h;
      float cs = mu_c + eps_c[eo] * sd_c;
      out[OFF_HID + ((size_t)b * S_ + t) * H_ + j] = hs;
      size_t mo = ((size_t)b * (S_ + 1) + (t + 1)) * H_ + j;
      out[OFF_HMU + mo] = mu_h;
      out[OFF_HSD + mo] = sd_h;
      out[OFF_CMU + mo] = mu_c;
      out[OFF_CSD + mo] = sd_c;
      h_bf[b * H_ + j] = (bf16_t)hs;   // carried h_s
      c_buf[b * H_ + j] = cs;          // carried c_s
      if (t == S_ - 1) {
        out[OFF_HT + (size_t)b * H_ + j] = hs;
        out[OFF_CT + (size_t)b * H_ + j] = cs;
      }
    }
    grid_barrier(bar);
  }
}

// ---------------- host launcher ----------------
extern "C" void kernel_launch(void* const* d_in, const int* in_sizes, int n_in,
                              void* d_out, int out_size, void* d_ws, size_t ws_size,
                              hipStream_t stream) {
  const float* x     = (const float*)d_in[0];
  const float* W_ih  = (const float*)d_in[1];
  const float* W_hh  = (const float*)d_in[2];
  const float* b_ih  = (const float*)d_in[3];
  const float* b_hh  = (const float*)d_in[4];
  const float* W_hg  = (const float*)d_in[5];
  const float* W_cg  = (const float*)d_in[6];
  const float* eps_h = (const float*)d_in[7];
  const float* eps_c = (const float*)d_in[8];
  float* out = (float*)d_out;

  char* ws = (char*)d_ws;
  size_t off = 0;
  auto alloc = [&](size_t bytes) -> void* {
    void* p = ws + off;
    off += (bytes + 255) & ~(size_t)255;
    return p;
  };
  bf16_t* xbf   = (bf16_t*)alloc((size_t)B_ * S_ * D_ * 2);   // 32 MB
  bf16_t* wih   = (bf16_t*)alloc((size_t)G4 * D_ * 2);        // 2 MB
  bf16_t* whh   = (bf16_t*)alloc((size_t)G4 * H_ * 2);        // 2 MB
  bf16_t* whgT  = (bf16_t*)alloc((size_t)NH2 * H_ * 2);       // 1 MB
  bf16_t* wcgT  = (bf16_t*)alloc((size_t)NH2 * H_ * 2);       // 1 MB
  float*  bias  = (float*)alloc((size_t)G4 * 4);
  float*  Gx    = (float*)alloc((size_t)B_ * S_ * G4 * 4);    // 256 MB
  bf16_t* h_bf  = (bf16_t*)alloc((size_t)B_ * H_ * 2);
  bf16_t* hn_bf = (bf16_t*)alloc((size_t)B_ * H_ * 2);
  bf16_t* cn_bf = (bf16_t*)alloc((size_t)B_ * H_ * 2);
  float*  c_buf = (float*)alloc((size_t)B_ * H_ * 4);
  unsigned* bar = (unsigned*)alloc(256);
  (void)in_sizes; (void)n_in; (void)out_size; (void)ws_size;

  // conversions
  {
    int n = B_ * S_ * D_;
    cvt_bf16_kernel<<<(n + 255) / 256, 256, 0, stream>>>(x, xbf, n);
    n = G4 * D_;
    cvt_bf16_kernel<<<(n + 255) / 256, 256, 0, stream>>>(W_ih, wih, n);
    cvt_bf16_kernel<<<(n + 255) / 256, 256, 0, stream>>>(W_hh, whh, n);
    n = NH2 * H_;
    cvt_T_kernel<<<(n + 255) / 256, 256, 0, stream>>>(W_hg, whgT);
    cvt_T_kernel<<<(n + 255) / 256, 256, 0, stream>>>(W_cg, wcgT);
    bias_kernel<<<(G4 + 255) / 256, 256, 0, stream>>>(b_ih, b_hh, bias);
  }
  // big parallel input projection (off the sequential critical path)
  gemm_gx_kernel<<<dim3(G4 / 128, (B_ * S_) / 64), 256, 0, stream>>>(xbf, wih, bias, Gx);
  // t = 0 state + barrier reset
  init_kernel<<<(B_ * H_ + 255) / 256, 256, 0, stream>>>(eps_h, eps_c, h_bf, c_buf, out, bar);
  // one persistent launch runs the whole 512-step recurrence
  recurrence_kernel<<<NWG, 256, LDS_TOTAL, stream>>>(whh, whgT, wcgT, Gx, eps_h, eps_c,
                                                     h_bf, hn_bf, cn_bf, c_buf, out, bar);
}
